// MultiFrequencyConformer_83811991814394
// MI455X (gfx1250) — matrix operand order repaired and checked by reading, not compiled, not run on hardware
//
#include <hip/hip_runtime.h>
#include <hip/hip_bf16.h>

// ---------------------------------------------------------------------------
// Types for CDNA5 WMMA (wave32): v_wmma_f32_16x16x32_bf16
// ---------------------------------------------------------------------------
typedef __bf16 bf16_t;
typedef __attribute__((ext_vector_type(4)))  __bf16 bf16x4;
typedef __attribute__((ext_vector_type(8)))  __bf16 bf16x8;
typedef __attribute__((ext_vector_type(16))) __bf16 bf16x16;
typedef __attribute__((ext_vector_type(8)))  float  floatx8;
typedef __attribute__((ext_vector_type(4)))  float  floatx4;

#define BM 128
#define BN 64
#define BK 32
#define KP (BK + 8)   // padded row: 40 bf16 = 80 B (16B aligned)

// Model constants
#define BATCH 16
#define SEQ   2048
#define DMOD  512
#define HEADS 8
#define HDIM  64
#define CS    256
#define NCH   8      // SEQ / CS
#define NCLS  10
#define NPAD  64     // classifier N padded to tile width
#define KCONV 31

struct GemmP {
    const float* A; long long sA0, sA1; int lda;
    const float* B; long long sB0, sB1; int ldb;
    float*       C; long long sC0, sC1; int ldc;
    const float* R; long long sR0, sR1; int ldr;   // residual (nullable)
    const float* bias;                              // length >= N (nullable)
    float*       Sst;                               // fused memstate S (nullable)
    float alpha, beta;
    int M, N, K, h_per;
};

// ---------------------------------------------------------------------------
// Batched GEMM: C = alpha * op(A) @ op(B) + bias + beta * R
// Fused-memstate mode (Sst != null):
//   g = alpha * op(A)@op(B); s = 0.9*S - 0.1*g; S = s; C = 0.99*C + s
// f32 in/out, bf16 WMMA with f32 accumulate.
// 256 threads (8 waves) compute a 128x64 tile; wave w owns rows
// [w*16, w*16+16) and all 64 cols -> 4 WMMAs per K-step, A-fragment reuse x4.
// Double-buffered LDS (one barrier per K-step); A staged [m][k], B staged
// transposed [n][k] so every WMMA fragment is two contiguous ds_load_b128.
// ALL launches must be tile-aligned: M%128==0, N%64==0, K%32==0.
// ---------------------------------------------------------------------------
template <int TA, int TB>
__global__ __launch_bounds__(256) void gemm_bf16_wmma(GemmP p) {
    __shared__ __align__(16) bf16_t As[2][BM][KP];   // [buf][m][k]
    __shared__ __align__(16) bf16_t Bt[2][BN][KP];   // [buf][n][k]

    const int z  = blockIdx.z;
    const int b0 = z / p.h_per;
    const int b1 = z % p.h_per;
    const float* Ab = p.A + b0 * p.sA0 + b1 * p.sA1;
    const float* Bb = p.B + b0 * p.sB0 + b1 * p.sB1;
    float*       Cb = p.C + b0 * p.sC0 + b1 * p.sC1;
    const float* Rb = p.R ? (p.R + b0 * p.sR0 + b1 * p.sR1) : nullptr;

    const int row0 = blockIdx.y * BM;
    const int col0 = blockIdx.x * BN;
    const int tid  = threadIdx.x;
    const int lane = tid & 31;
    const int wave = tid >> 5;                // 0..7 -> 16-row strip

    // per-thread staging register tiles
    floatx4 ra[4], rb[2];

    auto loadA = [&](int kt) {
        const int k0 = kt * BK;
        if (TA == 0) {
#pragma unroll
            for (int i = 0; i < 4; ++i) {     // 128x32 = 1024 float4
                int f = tid + i * 256;
                int m = f >> 3, k = (f & 7) * 4;
                ra[i] = *(const floatx4*)(Ab + (long long)(row0 + m) * p.lda + (k0 + k));
            }
        } else {                               // A stored [K,M], fast axis m
#pragma unroll
            for (int i = 0; i < 4; ++i) {
                int f = tid + i * 256;
                int k = f >> 5, m = (f & 31) * 4;
                ra[i] = *(const floatx4*)(Ab + (long long)(k0 + k) * p.lda + (row0 + m));
            }
        }
    };
    auto loadB = [&](int kt) {
        const int k0 = kt * BK;
        if (TB == 1) {                         // B stored [N,K], fast axis k
#pragma unroll
            for (int i = 0; i < 2; ++i) {      // 64x32 = 512 float4
                int f = tid + i * 256;
                int n = f >> 3, k = (f & 7) * 4;
                rb[i] = *(const floatx4*)(Bb + (long long)(col0 + n) * p.ldb + (k0 + k));
            }
        } else {                               // B stored [K,N], fast axis n
#pragma unroll
            for (int i = 0; i < 2; ++i) {
                int f = tid + i * 256;
                int k = f >> 4, n = (f & 15) * 4;
                rb[i] = *(const floatx4*)(Bb + (long long)(k0 + k) * p.ldb + (col0 + n));
            }
        }
    };
    auto storeA = [&](int buf) {
        if (TA == 0) {
#pragma unroll
            for (int i = 0; i < 4; ++i) {
                int f = tid + i * 256;
                int m = f >> 3, k = (f & 7) * 4;
                bf16x4 t;
#pragma unroll
                for (int j = 0; j < 4; ++j) t[j] = (bf16_t)ra[i][j];
                *(bf16x4*)&As[buf][m][k] = t;           // 8B ds_store
            }
        } else {
#pragma unroll
            for (int i = 0; i < 4; ++i) {
                int f = tid + i * 256;
                int k = f >> 5, m = (f & 31) * 4;
#pragma unroll
                for (int j = 0; j < 4; ++j) As[buf][m + j][k] = (bf16_t)ra[i][j];
            }
        }
    };
    auto storeB = [&](int buf) {
        if (TB == 1) {
#pragma unroll
            for (int i = 0; i < 2; ++i) {
                int f = tid + i * 256;
                int n = f >> 3, k = (f & 7) * 4;
                bf16x4 t;
#pragma unroll
                for (int j = 0; j < 4; ++j) t[j] = (bf16_t)rb[i][j];
                *(bf16x4*)&Bt[buf][n][k] = t;
            }
        } else {
#pragma unroll
            for (int i = 0; i < 2; ++i) {
                int f = tid + i * 256;
                int k = f >> 4, n = (f & 15) * 4;
#pragma unroll
                for (int j = 0; j < 4; ++j) Bt[buf][n + j][k] = (bf16_t)rb[i][j];
            }
        }
    };

    floatx8 acc0 = {}, acc1 = {}, acc2 = {}, acc3 = {};
    const int nk = p.K >> 5;                   // K % 32 == 0 by construction

    // prologue: stage tile 0 into buffer 0
    loadA(0); loadB(0);
    storeA(0); storeB(0);
    __syncthreads();

    int buf = 0;
    for (int kt = 0; kt < nk; ++kt) {
        const bool more = (kt + 1) < nk;
        if (more) {
            loadA(kt + 1); loadB(kt + 1);      // overlap with WMMA below
            if (kt + 2 < nk) {                  // L2 prefetch two tiles ahead
                const int kp = (kt + 2) * BK;
                __builtin_prefetch(Ab + (long long)(row0 + (tid >> 3)) * p.lda + kp, 0, 0);
                __builtin_prefetch(Bb + (long long)kp * p.ldb + col0, 0, 0);
            }
        }

        // ---- fragments: two ds_load_b128 each ----
        const int mloc = wave * 16 + (lane & 15);
        const int ahi  = (lane >> 4) * 8;
        bf16x8 a0 = *(const bf16x8*)&As[buf][mloc][ahi];
        bf16x8 a1 = *(const bf16x8*)&As[buf][mloc][16 + ahi];
        bf16x16 afrag = __builtin_shufflevector(a0, a1,
            0, 1, 2, 3, 4, 5, 6, 7, 8, 9, 10, 11, 12, 13, 14, 15);

        const int kb = (lane >> 4) * 8;
        const int nl = lane & 15;
#pragma unroll
        for (int c = 0; c < 4; ++c) {
            bf16x8 bl = *(const bf16x8*)&Bt[buf][c * 16 + nl][kb];
            bf16x8 bh = *(const bf16x8*)&Bt[buf][c * 16 + nl][16 + kb];
            bf16x16 bfrag = __builtin_shufflevector(bl, bh,
                0, 1, 2, 3, 4, 5, 6, 7, 8, 9, 10, 11, 12, 13, 14, 15);
            if (c == 0) acc0 = __builtin_amdgcn_wmma_f32_16x16x32_bf16(
                            false, afrag, false, bfrag, (short)0, acc0, false, false);
            if (c == 1) acc1 = __builtin_amdgcn_wmma_f32_16x16x32_bf16(
                            false, afrag, false, bfrag, (short)0, acc1, false, false);
            if (c == 2) acc2 = __builtin_amdgcn_wmma_f32_16x16x32_bf16(
                            false, afrag, false, bfrag, (short)0, acc2, false, false);
            if (c == 3) acc3 = __builtin_amdgcn_wmma_f32_16x16x32_bf16(
                            false, afrag, false, bfrag, (short)0, acc3, false, false);
        }

        if (more) { storeA(buf ^ 1); storeB(buf ^ 1); }
        __syncthreads();
        buf ^= 1;
    }

    // ---- epilogue: D VGPR r -> M = r + 8*lanehi, N = lane&15 ----
    const int nl = lane & 15;
    const int mb = row0 + wave * 16 + ((lane >> 4) * 8);
    if (p.Sst) {
        // fused memory-state EMA: C plays the role of M-state, same layout
        float* Sb = p.Sst + b0 * p.sC0 + b1 * p.sC1;
#pragma unroll
        for (int c = 0; c < 4; ++c) {
            const int gn = col0 + c * 16 + nl;
            const floatx8& ac = (c == 0) ? acc0 : (c == 1) ? acc1 : (c == 2) ? acc2 : acc3;
#pragma unroll
            for (int r = 0; r < 8; ++r) {
                long long idx = (long long)(mb + r) * p.ldc + gn;
                float g = p.alpha * ac[r];
                float s = 0.9f * Sb[idx] - 0.1f * g;
                Sb[idx] = s;
                Cb[idx] = 0.99f * Cb[idx] + s;
            }
        }
    } else {
#pragma unroll
        for (int c = 0; c < 4; ++c) {
            const int gn = col0 + c * 16 + nl;
            const float bv = p.bias ? p.bias[gn] : 0.f;
            const floatx8& ac = (c == 0) ? acc0 : (c == 1) ? acc1 : (c == 2) ? acc2 : acc3;
#pragma unroll
            for (int r = 0; r < 8; ++r) {
                long long gm = mb + r;
                float v = p.alpha * ac[r] + bv;
                if (Rb) v += p.beta * Rb[gm * p.ldr + gn];
                Cb[gm * p.ldc + gn] = v;
            }
        }
    }
}

// ---------------------------------------------------------------------------
// LayerNorm over D=512, one block (256 threads) per row, 2 elems/thread.
// ---------------------------------------------------------------------------
__global__ __launch_bounds__(256) void layernorm512(
    const float* __restrict__ x, long long xbs,
    float* __restrict__ y, long long ybs, int rowsPerBatch,
    const float* __restrict__ g, const float* __restrict__ b) {
    __shared__ float red[256];
    int row = blockIdx.x;
    int bi = row / rowsPerBatch, l = row % rowsPerBatch;
    const float* xr = x + (long long)bi * xbs + (long long)l * DMOD;
    float*       yr = y + (long long)bi * ybs + (long long)l * DMOD;
    int t = threadIdx.x;
    float v0 = xr[t], v1 = xr[t + 256];
    red[t] = v0 + v1;
    __syncthreads();
    for (int s = 128; s > 0; s >>= 1) {
        if (t < s) red[t] += red[t + s];
        __syncthreads();
    }
    float mean = red[0] * (1.f / DMOD);
    __syncthreads();
    float d0 = v0 - mean, d1 = v1 - mean;
    red[t] = d0 * d0 + d1 * d1;
    __syncthreads();
    for (int s = 128; s > 0; s >>= 1) {
        if (t < s) red[t] += red[t + s];
        __syncthreads();
    }
    float inv = rsqrtf(red[0] * (1.f / DMOD) + 1e-5f);
    yr[t]       = d0 * inv * g[t]       + b[t];
    yr[t + 256] = d1 * inv * g[t + 256] + b[t + 256];
}

// ---------------------------------------------------------------------------
// Softmax over rows of length 256 (attention scores), one block per row.
// ---------------------------------------------------------------------------
__global__ __launch_bounds__(256) void softmax256(float* __restrict__ s) {
    __shared__ float red[256];
    float* r = s + (long long)blockIdx.x * 256;
    int t = threadIdx.x;
    float v = r[t];
    red[t] = v;
    __syncthreads();
    for (int k = 128; k > 0; k >>= 1) {
        if (t < k) red[t] = fmaxf(red[t], red[t + k]);
        __syncthreads();
    }
    float mx = red[0];
    __syncthreads();
    float e = __expf(v - mx);
    red[t] = e;
    __syncthreads();
    for (int k = 128; k > 0; k >>= 1) {
        if (t < k) red[t] += red[t + k];
        __syncthreads();
    }
    r[t] = e / red[0];
}

// ---------------------------------------------------------------------------
// GLU: y2d [B,CS,2D] -> out [B,CS,D], a * sigmoid(g)
// ---------------------------------------------------------------------------
__global__ void glu_k(const float* __restrict__ y2d, float* __restrict__ out,
                      long long n) {
    long long i = (long long)blockIdx.x * blockDim.x + threadIdx.x;
    if (i >= n) return;
    long long row = i / DMOD;
    int c = (int)(i % DMOD);
    float a = y2d[row * (2 * DMOD) + c];
    float g = y2d[row * (2 * DMOD) + DMOD + c];
    out[i] = a * (1.f / (1.f + __expf(-g)));
}

// ---------------------------------------------------------------------------
// Depthwise conv (K=31, SAME within chunk) + bias + swish.
// ---------------------------------------------------------------------------
__global__ void dwconv_swish(const float* __restrict__ in,
                             const float* __restrict__ w,
                             const float* __restrict__ bias,
                             float* __restrict__ out, long long n) {
    long long i = (long long)blockIdx.x * blockDim.x + threadIdx.x;
    if (i >= n) return;
    int c = (int)(i % DMOD);
    long long rl = i / DMOD;
    int l = (int)(rl % CS);
    long long b = rl / CS;
    const float* ib = in + b * (long long)(CS * DMOD);
    float acc = bias[c];
#pragma unroll
    for (int j = 0; j < KCONV; ++j) {
        int t = l + j - (KCONV / 2);
        if (t >= 0 && t < CS) acc += ib[(long long)t * DMOD + c] * w[j * DMOD + c];
    }
    out[i] = acc * (1.f / (1.f + __expf(-acc)));
}

// ---------------------------------------------------------------------------
// Classifier padding helpers: WcPad [64,512] (rows >= 10 zero), bcPad [64].
// ---------------------------------------------------------------------------
__global__ void pad_cls(const float* __restrict__ Wc, const float* __restrict__ bc,
                        float* __restrict__ WcPad, float* __restrict__ bcPad) {
    int i = blockIdx.x * blockDim.x + threadIdx.x;
    if (i < NPAD * DMOD) {
        int r = i / DMOD;
        WcPad[i] = (r < NCLS) ? Wc[i] : 0.f;
    }
    if (i < NPAD) bcPad[i] = (i < NCLS) ? bc[i] : 0.f;
}

__global__ void copy_out(const float* __restrict__ OutPad, float* __restrict__ out,
                         long long n) {
    long long i = (long long)blockIdx.x * blockDim.x + threadIdx.x;
    if (i >= n) return;
    long long r = i / NCLS;
    int c = (int)(i % NCLS);
    out[i] = OutPad[r * NPAD + c];
}

// ---------------------------------------------------------------------------
// Host-side launch helper (dispatch on transpose flags)
// ---------------------------------------------------------------------------
static inline void launch_gemm_full(hipStream_t st,
    const float* A, long long sA0, long long sA1, int lda, int tA,
    const float* Bp, long long sB0, long long sB1, int ldb, int tB,
    float* C, long long sC0, long long sC1, int ldc,
    const float* R, long long sR0, long long sR1, int ldr,
    const float* bias, float* Sst, float alpha, float beta,
    int M, int N, int K, int batches, int h_per) {
    GemmP p;
    p.A = A; p.sA0 = sA0; p.sA1 = sA1; p.lda = lda;
    p.B = Bp; p.sB0 = sB0; p.sB1 = sB1; p.ldb = ldb;
    p.C = C; p.sC0 = sC0; p.sC1 = sC1; p.ldc = ldc;
    p.R = R; p.sR0 = sR0; p.sR1 = sR1; p.ldr = ldr;
    p.bias = bias; p.Sst = Sst; p.alpha = alpha; p.beta = beta;
    p.M = M; p.N = N; p.K = K; p.h_per = h_per;
    dim3 grid(N / BN, M / BM, batches);
    dim3 blk(256);
    if (tA == 0 && tB == 0)      gemm_bf16_wmma<0, 0><<<grid, blk, 0, st>>>(p);
    else if (tA == 0 && tB == 1) gemm_bf16_wmma<0, 1><<<grid, blk, 0, st>>>(p);
    else if (tA == 1 && tB == 0) gemm_bf16_wmma<1, 0><<<grid, blk, 0, st>>>(p);
    else                         gemm_bf16_wmma<1, 1><<<grid, blk, 0, st>>>(p);
}

static inline void launch_gemm(hipStream_t st,
    const float* A, long long sA0, long long sA1, int lda, int tA,
    const float* Bp, long long sB0, long long sB1, int ldb, int tB,
    float* C, long long sC0, long long sC1, int ldc,
    const float* R, long long sR0, long long sR1, int ldr,
    const float* bias, float alpha, float beta,
    int M, int N, int K, int batches, int h_per) {
    launch_gemm_full(st, A, sA0, sA1, lda, tA, Bp, sB0, sB1, ldb, tB,
                     C, sC0, sC1, ldc, R, sR0, sR1, ldr, bias, nullptr,
                     alpha, beta, M, N, K, batches, h_per);
}

extern "C" void kernel_launch(void* const* d_in, const int* in_sizes, int n_in,
                              void* d_out, int out_size, void* d_ws, size_t ws_size,
                              hipStream_t stream) {
    // ----- inputs (setup_inputs order) -----
    const float* x     = (const float*)d_in[0];
    const float* Win   = (const float*)d_in[1];
    const float* b_in  = (const float*)d_in[2];
    const float* Wq_a  = (const float*)d_in[3];
    const float* bq_a  = (const float*)d_in[4];
    const float* Wk_a  = (const float*)d_in[5];
    const float* bk_a  = (const float*)d_in[6];
    const float* Wv_a  = (const float*)d_in[7];
    const float* bv_a  = (const float*)d_in[8];
    const float* Wo_a  = (const float*)d_in[9];
    const float* bo_a  = (const float*)d_in[10];
    const float* ln_g  = (const float*)d_in[11];
    const float* ln_b  = (const float*)d_in[12];
    const float* pw1_W = (const float*)d_in[13];
    const float* pw1_b = (const float*)d_in[14];
    const float* dw_W  = (const float*)d_in[15];
    const float* dw_b  = (const float*)d_in[16];
    const float* pw2_W = (const float*)d_in[17];
    const float* pw2_b = (const float*)d_in[18];
    const float* Wq_t  = (const float*)d_in[19];
    const float* Wk_t  = (const float*)d_in[20];
    const float* Wv_t  = (const float*)d_in[21];
    const float* Wq_d  = (const float*)d_in[22];
    const float* Wk_d  = (const float*)d_in[23];
    const float* Wv_d  = (const float*)d_in[24];
    const float* nrm_g = (const float*)d_in[25];
    const float* nrm_b = (const float*)d_in[26];
    const float* Wc    = (const float*)d_in[27];
    const float* bc    = (const float*)d_in[28];

    // ----- workspace layout (floats) -----
    float* ws = (float*)d_ws;
    size_t off = 0;
    auto alloc = [&](size_t n) { float* p = ws + off; off += n; return p; };
    const long long CD  = (long long)CS * DMOD;      // chunk tensor
    const long long TD  = (long long)SEQ * DMOD;     // per-batch h stride
    const long long DD  = (long long)DMOD * DMOD;

    float* H    = alloc((size_t)BATCH * SEQ * DMOD);         // activations (in-place)
    float* Qb   = alloc((size_t)BATCH * CD);
    float* Kb   = alloc((size_t)BATCH * CD);
    float* Vb   = alloc((size_t)BATCH * CD);
    float* T0   = alloc((size_t)BATCH * CD);
    float* T1   = alloc((size_t)BATCH * CD);
    float* Y2   = alloc((size_t)BATCH * CS * 2 * DMOD);      // also classifier OutPad
    float* SC   = alloc((size_t)BATCH * HEADS * CS * CS);    // attn scores
    float* Mt   = alloc((size_t)BATCH * DD);
    float* St   = alloc((size_t)BATCH * DD);
    float* Md   = alloc((size_t)BATCH * DD);
    float* Sd   = alloc((size_t)BATCH * DD);
    float* WcP  = alloc((size_t)NPAD * DMOD);                // padded classifier W
    float* bcP  = alloc((size_t)NPAD);                       // padded classifier b
    (void)ws_size; (void)in_sizes; (void)n_in; (void)out_size;

    // zero the four memory-state buffers (contiguous)
    hipMemsetAsync(Mt, 0, (size_t)4 * BATCH * DD * sizeof(float), stream);
    // build padded classifier weights
    pad_cls<<<(NPAD * DMOD + 255) / 256, 256, 0, stream>>>(Wc, bc, WcP, bcP);

    const long long nEl = (long long)BATCH * CD;             // chunk elem count
    const int eblk = 256;

    // ----- input projection: H = x @ Win^T + b_in  [B*T,128]->[B*T,512] -----
    launch_gemm(stream, x, 0, 0, 128, 0,  Win, 0, 0, 128, 1,
                H, 0, 0, DMOD,  nullptr, 0, 0, 0,  b_in, 1.f, 0.f,
                BATCH * SEQ, DMOD, 128, 1, 1);

    for (int n = 0; n < NCH; ++n) {
        float* Hc = H + (long long)n * CD;   // chunk view; per-batch stride TD

        // ===== mem_update (deep, Md/Sd) on xc =====
        launch_gemm(stream, Hc, TD, 0, DMOD, 0, Wq_d, 0, 0, DMOD, 1,
                    Qb, CD, 0, DMOD, nullptr, 0, 0, 0, nullptr, 1.f, 0.f,
                    CS, DMOD, DMOD, BATCH, 1);
        launch_gemm(stream, Hc, TD, 0, DMOD, 0, Wk_d, 0, 0, DMOD, 1,
                    Kb, CD, 0, DMOD, nullptr, 0, 0, 0, nullptr, 1.f, 0.f,
                    CS, DMOD, DMOD, BATCH, 1);
        launch_gemm(stream, Hc, TD, 0, DMOD, 0, Wv_d, 0, 0, DMOD, 1,
                    Vb, CD, 0, DMOD, nullptr, 0, 0, 0, nullptr, 1.f, 0.f,
                    CS, DMOD, DMOD, BATCH, 1);
        // pmv = k @ Md - v   (old Md)
        launch_gemm(stream, Kb, CD, 0, DMOD, 0, Md, DD, 0, DMOD, 0,
                    T0, CD, 0, DMOD, Vb, CD, 0, DMOD, nullptr, 1.f, -1.f,
                    CS, DMOD, DMOD, BATCH, 1);
        // xc += 0.5 * q @ Md (old Md) -- must precede the fused state update
        launch_gemm(stream, Qb, CD, 0, DMOD, 0, Md, DD, 0, DMOD, 0,
                    Hc, TD, 0, DMOD, Hc, TD, 0, DMOD, nullptr, 0.5f, 1.f,
                    CS, DMOD, DMOD, BATCH, 1);
        // fused: g = (1/CS) k^T @ pmv ; Sd = .9 Sd - .1 g ; Md = .99 Md + Sd
        launch_gemm_full(stream, Kb, CD, 0, DMOD, 1, T0, CD, 0, DMOD, 0,
                    Md, DD, 0, DMOD, nullptr, 0, 0, 0, nullptr, Sd, 1.f / CS, 0.f,
                    DMOD, DMOD, CS, BATCH, 1);

        // ===== attention =====
        launch_gemm(stream, Hc, TD, 0, DMOD, 0, Wq_a, 0, 0, DMOD, 1,
                    Qb, CD, 0, DMOD, nullptr, 0, 0, 0, bq_a, 1.f, 0.f,
                    CS, DMOD, DMOD, BATCH, 1);
        launch_gemm(stream, Hc, TD, 0, DMOD, 0, Wk_a, 0, 0, DMOD, 1,
                    Kb, CD, 0, DMOD, nullptr, 0, 0, 0, bk_a, 1.f, 0.f,
                    CS, DMOD, DMOD, BATCH, 1);
        launch_gemm(stream, Hc, TD, 0, DMOD, 0, Wv_a, 0, 0, DMOD, 1,
                    Vb, CD, 0, DMOD, nullptr, 0, 0, 0, bv_a, 1.f, 0.f,
                    CS, DMOD, DMOD, BATCH, 1);
        // scores = (1/8) q @ k^T  per (b,h)
        launch_gemm(stream, Qb, CD, HDIM, DMOD, 0, Kb, CD, HDIM, DMOD, 1,
                    SC, (long long)HEADS * CS * CS, (long long)CS * CS, CS,
                    nullptr, 0, 0, 0, nullptr, 0.125f, 0.f,
                    CS, CS, HDIM, BATCH * HEADS, HEADS);
        softmax256<<<BATCH * HEADS * CS, 256, 0, stream>>>(SC);
        // o = a @ v  per (b,h), written head-interleaved into T0
        launch_gemm(stream, SC, (long long)HEADS * CS * CS, (long long)CS * CS, CS, 0,
                    Vb, CD, HDIM, DMOD, 0,
                    T0, CD, HDIM, DMOD, nullptr, 0, 0, 0, nullptr, 1.f, 0.f,
                    CS, HDIM, CS, BATCH * HEADS, HEADS);
        // xc += o @ Wo^T + bo
        launch_gemm(stream, T0, CD, 0, DMOD, 0, Wo_a, 0, 0, DMOD, 1,
                    Hc, TD, 0, DMOD, Hc, TD, 0, DMOD, bo_a, 1.f, 1.f,
                    CS, DMOD, DMOD, BATCH, 1);

        // ===== conv module =====
        layernorm512<<<BATCH * CS, 256, 0, stream>>>(Hc, TD, T0, CD, CS, ln_g, ln_b);
        launch_gemm(stream, T0, CD, 0, DMOD, 0, pw1_W, 0, 0, DMOD, 1,
                    Y2, (long long)CS * 2 * DMOD, 0, 2 * DMOD,
                    nullptr, 0, 0, 0, pw1_b, 1.f, 0.f,
                    CS, 2 * DMOD, DMOD, BATCH, 1);
        glu_k<<<(int)((nEl + eblk - 1) / eblk), eblk, 0, stream>>>(Y2, T1, nEl);
        dwconv_swish<<<(int)((nEl + eblk - 1) / eblk), eblk, 0, stream>>>(T1, dw_W, dw_b, T0, nEl);
        launch_gemm(stream, T0, CD, 0, DMOD, 0, pw2_W, 0, 0, DMOD, 1,
                    Hc, TD, 0, DMOD, Hc, TD, 0, DMOD, pw2_b, 1.f, 1.f,
                    CS, DMOD, DMOD, BATCH, 1);

        // ===== mem_update (token, Mt/St) on updated xc =====
        launch_gemm(stream, Hc, TD, 0, DMOD, 0, Wq_t, 0, 0, DMOD, 1,
                    Qb, CD, 0, DMOD, nullptr, 0, 0, 0, nullptr, 1.f, 0.f,
                    CS, DMOD, DMOD, BATCH, 1);
        launch_gemm(stream, Hc, TD, 0, DMOD, 0, Wk_t, 0, 0, DMOD, 1,
                    Kb, CD, 0, DMOD, nullptr, 0, 0, 0, nullptr, 1.f, 0.f,
                    CS, DMOD, DMOD, BATCH, 1);
        launch_gemm(stream, Hc, TD, 0, DMOD, 0, Wv_t, 0, 0, DMOD, 1,
                    Vb, CD, 0, DMOD, nullptr, 0, 0, 0, nullptr, 1.f, 0.f,
                    CS, DMOD, DMOD, BATCH, 1);
        launch_gemm(stream, Kb, CD, 0, DMOD, 0, Mt, DD, 0, DMOD, 0,
                    T0, CD, 0, DMOD, Vb, CD, 0, DMOD, nullptr, 1.f, -1.f,
                    CS, DMOD, DMOD, BATCH, 1);
        launch_gemm(stream, Qb, CD, 0, DMOD, 0, Mt, DD, 0, DMOD, 0,
                    Hc, TD, 0, DMOD, Hc, TD, 0, DMOD, nullptr, 0.5f, 1.f,
                    CS, DMOD, DMOD, BATCH, 1);
        launch_gemm_full(stream, Kb, CD, 0, DMOD, 1, T0, CD, 0, DMOD, 0,
                    Mt, DD, 0, DMOD, nullptr, 0, 0, 0, nullptr, St, 1.f / CS, 0.f,
                    DMOD, DMOD, CS, BATCH, 1);

        // ===== output layernorm (in place on chunk) =====
        layernorm512<<<BATCH * CS, 256, 0, stream>>>(Hc, TD, Hc, TD, CS, nrm_g, nrm_b);
    }

    // ----- classifier (padded N=64): Y2 = H @ WcP^T + bcP, then narrow copy -----
    launch_gemm(stream, H, 0, 0, DMOD, 0, WcP, 0, 0, DMOD, 1,
                Y2, 0, 0, NPAD, nullptr, 0, 0, 0, bcP, 1.f, 0.f,
                BATCH * SEQ, NPAD, DMOD, 1, 1);
    const long long nOut = (long long)BATCH * SEQ * NCLS;
    copy_out<<<(int)((nOut + 255) / 256), 256, 0, stream>>>(Y2, (float*)d_out, nOut);
}
